// GraphAttentionTransformerMD17_16183436772002
// MI455X (gfx1250) — matrix operand order repaired and checked by reading, not compiled, not run on hardware
//
#include <hip/hip_runtime.h>
#include <hip/hip_bf16.h>

typedef __attribute__((ext_vector_type(16))) _Float16 v16h;
typedef __attribute__((ext_vector_type(8)))  _Float16 v8h;
typedef __attribute__((ext_vector_type(8)))  float    v8f;

#define N_NODES 50000
#define N_EDGES 800000
#define FEAT    144

__device__ __forceinline__ v8f wmma16(v16h a, v16h b, v8f c) {
  // D(f32 16x16) = A(f16 16x32) * B(f16 32x16) + C
  return __builtin_amdgcn_wmma_f32_16x16x32_f16(false, a, false, b, (short)0, c,
                                                false, false);
}
__device__ __forceinline__ float silu(float x) { return x / (1.0f + __expf(-x)); }
__device__ __forceinline__ v16h joinh(v8h lo, v8h hi) {
  return __builtin_shufflevector(lo, hi, 0,1,2,3,4,5,6,7,8,9,10,11,12,13,14,15);
}
__device__ __forceinline__ v8f zero8f() {
  v8f z;
#pragma unroll
  for (int i = 0; i < 8; i++) z[i] = 0.0f;
  return z;
}
__device__ __forceinline__ v8h zero8h() {
  v8h z;
#pragma unroll
  for (int i = 0; i < 8; i++) z[i] = (_Float16)0.0f;
  return z;
}

// ---------------------------------------------------------------------------
// Pack a row-major fp32 weight matrix W[K][Nout] into the per-lane WMMA
// B-fragment layout (f16). Per ISA 7.12.2 (16-bit B, 32x16): lane n (0..15)
// holds column n with K=0..7,16..23 packed along the fragment; lanes 16..31 hold
// K=8..15,24..31. Fragment for (kb,nt) of lane L is 16 contiguous halves.
// ---------------------------------------------------------------------------
__global__ void pack_wmma_b(const float* __restrict__ W, _Float16* __restrict__ dst,
                            int Ksrc, int KB, int NT, int Nout) {
  int total = KB * NT * 32 * 16;
  for (int idx = blockIdx.x * blockDim.x + threadIdx.x; idx < total;
       idx += gridDim.x * blockDim.x) {
    int i    = idx & 15;
    int lane = (idx >> 4) & 31;
    int blk  = idx >> 9;
    int nt   = blk % NT;
    int kb   = blk / NT;
    int k = kb * 32 + ((i >> 3) << 4) + ((lane >> 4) << 3) + (i & 7);   // B-matrix K mapping
    int n = nt * 16 + (lane & 15);
    float v = (k < Ksrc) ? W[(size_t)k * Nout + n] : 0.0f;
    dst[idx] = (_Float16)v;
  }
}

// A-matrix (16x32 f16) K index for element i of lane: per ISA 7.12.2
// lanes 0-15: K = {0..7,16..23}; lanes 16-31: K = {8..15,24..31}
__device__ __forceinline__ int kmapA(int lh, int i) {
  return ((i >> 3) << 4) + (lh << 3) + (i & 7);
}

// ---------------------------------------------------------------------------
// Edge kernel: one wave per 16-edge tile.
//   rbf[16,128] (analytic, straight into A fragments)
//   -> WMMA 128->64 (silu) -> WMMA 64->64 (silu) -> WMMA 64->16
//   -> outer with sh[9] -> f32 atomic scatter into deg_embed[N,144]
// ---------------------------------------------------------------------------
__global__ __launch_bounds__(256) void edge_kernel(
    const float* __restrict__ pos, const int* __restrict__ esrc,
    const int* __restrict__ edst, const float* __restrict__ b1,
    const float* __restrict__ b2, const _Float16* __restrict__ wpack,
    float* __restrict__ deg) {
  __shared__ __align__(32) _Float16 sW[13312];     // W1p(8192) W2p(4096) W3p(1024)
  __shared__ __align__(32) _Float16 sH[8][16 * 72];
  __shared__ float sB1[64], sB2[64];
  __shared__ float sSH[8][16 * 10];
  __shared__ int   sDST[8][16];

  const int tid = threadIdx.x;
  for (int i = tid; i < 13312; i += 256) sW[i] = wpack[i];
  if (tid < 64) { sB1[tid] = b1[tid]; sB2[tid] = b2[tid]; }
  __syncthreads();

  const int lane = tid & 31;
  const int wv   = tid >> 5;
  const int row  = lane & 15;
  const int lh   = lane >> 4;
  const int gwave = (int)((blockIdx.x * blockDim.x + tid) >> 5);
  const int nwave = (int)((gridDim.x * blockDim.x) >> 5);
  _Float16* h = &sH[wv][0];

  const float kStart = 0.006737946999085467f;               // exp(-5)
  const float kStep  = (1.0f - kStart) * (1.0f / 127.0f);
  const float tbw    = (2.0f / 128.0f) * (1.0f - kStart);
  const float kBeta  = 1.0f / (tbw * tbw);
  const float invSqrtDeg = rsqrtf(15.57930850982666f);

  const int ntiles = N_EDGES / 16;
  for (int tile = gwave; tile < ntiles; tile += nwave) {
    const int e0 = tile * 16;
    const int e  = e0 + row;
    const int vs = esrc[e], vd = edst[e];
    const float ex = pos[3 * vs + 0] - pos[3 * vd + 0];
    const float ey = pos[3 * vs + 1] - pos[3 * vd + 1];
    const float ez = pos[3 * vs + 2] - pos[3 * vd + 2];
    const float d  = sqrtf(ex * ex + ey * ey + ez * ez);
    const float texp = __expf(-d);
    const float cut =
        (d < 5.0f) ? 0.5f * (__cosf(d * 0.6283185307179586f) + 1.0f) : 0.0f;

    if (lh == 0) {  // stage sh + dst once per edge row
      const float ir = 1.0f / fmaxf(d, 1e-12f);
      const float x = ex * ir, y = ey * ir, z = ez * ir;
      const float s3 = 1.7320508075688772f, s5 = 2.23606797749979f,
                  s15 = 3.872983346207417f;
      float* sp = &sSH[wv][row * 10];
      sp[0] = 1.0f; sp[1] = s3 * x; sp[2] = s3 * y; sp[3] = s3 * z;
      sp[4] = s15 * x * z; sp[5] = s15 * x * y;
      sp[6] = s5 * (y * y - 0.5f * (x * x + z * z));
      sp[7] = s15 * y * z; sp[8] = 0.5f * s15 * (z * z - x * x);
      sDST[wv][row] = vd;
    }

    // ---- layer 1: rbf A fragments computed analytically ----
    v16h a1[4];
#pragma unroll
    for (int kb = 0; kb < 4; kb++) {
#pragma unroll
      for (int i = 0; i < 16; i++) {
        int k = kb * 32 + kmapA(lh, i);
        float t = texp - (kStart + (float)k * kStep);
        a1[kb][i] = (_Float16)(cut * __expf(-kBeta * t * t));
      }
    }
    v8f acc[4];
#pragma unroll
    for (int nt = 0; nt < 4; nt++) acc[nt] = zero8f();
#pragma unroll
    for (int nt = 0; nt < 4; nt++)
#pragma unroll
      for (int kb = 0; kb < 4; kb++) {
        v16h b = *(const v16h*)&sW[((kb * 4 + nt) * 32 + lane) * 16];
        acc[nt] = wmma16(a1[kb], b, acc[nt]);
      }
#pragma unroll
    for (int nt = 0; nt < 4; nt++) {
      float bb = sB1[nt * 16 + row];
#pragma unroll
      for (int r = 0; r < 8; r++)
        h[(r + lh * 8) * 72 + nt * 16 + row] = (_Float16)silu(acc[nt][r] + bb);
    }
    __builtin_amdgcn_wave_barrier();

    // ---- layer 2: 64 -> 64 ----
    v16h a2[2];
#pragma unroll
    for (int kb = 0; kb < 2; kb++) {
      v8h lo = *(const v8h*)&h[row * 72 + kb * 32 + lh * 8];
      v8h hi = *(const v8h*)&h[row * 72 + kb * 32 + lh * 8 + 16];
      a2[kb] = joinh(lo, hi);
    }
    v8f acc2[4];
#pragma unroll
    for (int nt = 0; nt < 4; nt++) acc2[nt] = zero8f();
#pragma unroll
    for (int nt = 0; nt < 4; nt++)
#pragma unroll
      for (int kb = 0; kb < 2; kb++) {
        v16h b = *(const v16h*)&sW[8192 + ((kb * 4 + nt) * 32 + lane) * 16];
        acc2[nt] = wmma16(a2[kb], b, acc2[nt]);
      }
    __builtin_amdgcn_wave_barrier();
#pragma unroll
    for (int nt = 0; nt < 4; nt++) {
      float bb = sB2[nt * 16 + row];
#pragma unroll
      for (int r = 0; r < 8; r++)
        h[(r + lh * 8) * 72 + nt * 16 + row] = (_Float16)silu(acc2[nt][r] + bb);
    }
    __builtin_amdgcn_wave_barrier();

    // ---- layer 3: 64 -> 16 (no bias) ----
    v16h a3[2];
#pragma unroll
    for (int kb = 0; kb < 2; kb++) {
      v8h lo = *(const v8h*)&h[row * 72 + kb * 32 + lh * 8];
      v8h hi = *(const v8h*)&h[row * 72 + kb * 32 + lh * 8 + 16];
      a3[kb] = joinh(lo, hi);
    }
    v8f acc3 = zero8f();
#pragma unroll
    for (int kb = 0; kb < 2; kb++) {
      v16h b = *(const v16h*)&sW[12288 + (kb * 32 + lane) * 16];
      acc3 = wmma16(a3[kb], b, acc3);
    }

    // ---- scatter: lane owns channel `row`, 8 edge rows ----
#pragma unroll
    for (int r = 0; r < 8; r++) {
      int m = r + lh * 8;
      float rv = acc3[r] * invSqrtDeg;
      int dn = sDST[wv][m];
      float* base = deg + (size_t)dn * FEAT + row * 9;
      const float* sp = &sSH[wv][m * 10];
#pragma unroll
      for (int j = 0; j < 9; j++) unsafeAtomicAdd(base + j, rv * sp[j]);
    }
    __builtin_amdgcn_wave_barrier();
  }
}

// ---------------------------------------------------------------------------
// Node kernel: one wave per 16-node tile.
//   nf = atom_table[atom] + deg  -> WMMA [16,144]x[144,144] (K padded to 160)
//   -> silu -> dot with Wh2[144] + bh2 -> atomic into out[batch]
// ---------------------------------------------------------------------------
__global__ __launch_bounds__(128) void node_kernel(
    const float* __restrict__ atom_table, const int* __restrict__ node_atom,
    const float* __restrict__ deg, const _Float16* __restrict__ wh1p,
    const float* __restrict__ bh1, const float* __restrict__ wh2,
    const float* __restrict__ bh2, const int* __restrict__ batch,
    float* __restrict__ out) {
  __shared__ __align__(32) _Float16 sNF[4][16 * FEAT];
  __shared__ float sBH1[FEAT], sWH2[FEAT];

  const int tid = threadIdx.x;
  for (int i = tid; i < FEAT; i += 128) { sBH1[i] = bh1[i]; sWH2[i] = wh2[i]; }
  __syncthreads();

  const int lane = tid & 31;
  const int wv   = tid >> 5;
  const int row  = lane & 15;
  const int lh   = lane >> 4;
  const int gwave = (int)((blockIdx.x * blockDim.x + tid) >> 5);
  const int nwave = (int)((gridDim.x * blockDim.x) >> 5);
  const float invSqrtN = rsqrtf(18.03065905448718f);
  const float bh2v = bh2[0];
  _Float16* nf = &sNF[wv][0];

  const int ntiles = N_NODES / 16;
  for (int tile = gwave; tile < ntiles; tile += nwave) {
    const int n0 = tile * 16;
    {  // stage node features (f16); lane covers half a row
      const int node = n0 + row;
      const int at = node_atom[node];
      const float* arow = atom_table + (size_t)at * FEAT;
      const float* drow = deg + (size_t)node * FEAT;
      for (int k = lh * 72; k < lh * 72 + 72; k++)
        nf[row * FEAT + k] = (_Float16)(arow[k] + drow[k]);
    }
    __builtin_amdgcn_wave_barrier();

    v16h a[5];
#pragma unroll
    for (int kb = 0; kb < 5; kb++) {
      v8h lo = *(const v8h*)&nf[row * FEAT + kb * 32 + lh * 8];
      v8h hi = (kb == 4) ? zero8h()   // K=144..159 zero padding
                         : *(const v8h*)&nf[row * FEAT + kb * 32 + lh * 8 + 16];
      a[kb] = joinh(lo, hi);
    }
    v8f acc[9];
#pragma unroll
    for (int nt = 0; nt < 9; nt++) acc[nt] = zero8f();
#pragma unroll
    for (int nt = 0; nt < 9; nt++)
#pragma unroll
      for (int kb = 0; kb < 5; kb++) {
        v16h b = *(const v16h*)&wh1p[(size_t)((kb * 9 + nt) * 32 + lane) * 16];
        acc[nt] = wmma16(a[kb], b, acc[nt]);
      }
    __builtin_amdgcn_wave_barrier();
#pragma unroll
    for (int nt = 0; nt < 9; nt++) {
      float bb = sBH1[nt * 16 + row];
#pragma unroll
      for (int r = 0; r < 8; r++)
        nf[(r + lh * 8) * FEAT + nt * 16 + row] = (_Float16)silu(acc[nt][r] + bb);
    }
    __builtin_amdgcn_wave_barrier();

    float p = 0.0f;
    for (int k = lh * 72; k < lh * 72 + 72; k++)
      p += (float)nf[row * FEAT + k] * sWH2[k];
    p += __shfl_down(p, 16, 32);
    if (lane < 16) {
      const int node = n0 + row;
      unsafeAtomicAdd(&out[batch[node]], (p + bh2v) * invSqrtN);
    }
    __builtin_amdgcn_wave_barrier();
  }
}

extern "C" void kernel_launch(void* const* d_in, const int* in_sizes, int n_in,
                              void* d_out, int out_size, void* d_ws,
                              size_t ws_size, hipStream_t stream) {
  const float* pos        = (const float*)d_in[0];
  const float* atom_table = (const float*)d_in[1];
  const float* W1  = (const float*)d_in[2];
  const float* b1  = (const float*)d_in[3];
  const float* W2  = (const float*)d_in[4];
  const float* b2  = (const float*)d_in[5];
  const float* W3  = (const float*)d_in[6];
  const float* Wh1 = (const float*)d_in[7];
  const float* bh1 = (const float*)d_in[8];
  const float* Wh2 = (const float*)d_in[9];
  const float* bh2 = (const float*)d_in[10];
  const int* node_atom = (const int*)d_in[11];
  const int* edge_src  = (const int*)d_in[12];
  const int* edge_dst  = (const int*)d_in[13];
  const int* batch     = (const int*)d_in[14];

  char* ws = (char*)d_ws;
  float* deg = (float*)ws;                                   // [N,144] f32
  size_t off = (size_t)N_NODES * FEAT * sizeof(float);
  off = (off + 255) & ~(size_t)255;
  _Float16* w1p = (_Float16*)(ws + off);                     // 8192 halves
  _Float16* w2p = w1p + 8192;                                // 4096
  _Float16* w3p = w2p + 4096;                                // 1024
  size_t off2 = (off + 13312 * sizeof(_Float16) + 255) & ~(size_t)255;
  _Float16* wh1p = (_Float16*)(ws + off2);                   // 23040 halves

  // deterministic per-launch zeroing of accumulation buffers
  hipMemsetAsync(deg, 0, (size_t)N_NODES * FEAT * sizeof(float), stream);
  hipMemsetAsync(d_out, 0, (size_t)out_size * sizeof(float), stream);

  // pack fp32 weights into f16 WMMA B-fragment layouts
  pack_wmma_b<<<32, 256, 0, stream>>>(W1, w1p, 128, 4, 4, 64);
  pack_wmma_b<<<16, 256, 0, stream>>>(W2, w2p, 64, 2, 4, 64);
  pack_wmma_b<<<4, 256, 0, stream>>>(W3, w3p, 64, 2, 1, 16);
  pack_wmma_b<<<64, 256, 0, stream>>>(Wh1, wh1p, 144, 5, 9, 144);

  edge_kernel<<<1024, 256, 0, stream>>>(pos, edge_src, edge_dst, b1, b2, w1p, deg);
  node_kernel<<<512, 128, 0, stream>>>(atom_table, node_atom, deg, wh1p, bh1,
                                       Wh2, bh2, batch, (float*)d_out);
}
